// CPCModelJAX_87024627352063
// MI455X (gfx1250) — matrix operand order repaired and checked by reading, not compiled, not run on hardware
//
#include <hip/hip_runtime.h>
#include <hip/hip_bf16.h>

typedef __attribute__((ext_vector_type(16))) _Float16 v16h;
typedef __attribute__((ext_vector_type(8)))  float    v8f;

#define CPC_B     16
#define CPC_T     512
#define CPC_DIN   128
#define CPC_HENC  512
#define CPC_DLAT  256
#define CPC_HCTX  512
#define CPC_HPRED 512
#define CPC_K     12
#define CPC_N     (CPC_B * (CPC_T - CPC_K))   // 8000
#define Z_ELEMS   (CPC_B * CPC_T * CPC_DLAT)  // 2097152
#define C_ELEMS   (CPC_B * CPC_T * CPC_HCTX)  // 4194304

// ---------------------------------------------------------------------------
// WMMA fragment helpers (CDNA5 16x16x32 f16 layouts, wave32)
// A (16x32 f16): lane m = l%16; K = (l<16?0:16)+e (contiguous, same order as B)
// B (32x16 f16): lane n = l%16; K = (l<16?0:16)+e  -> contiguous when B is
//                stored transposed [N x K]
// C/D (16x16 f32): lane n = l%16; row = v + (l<16?0:8)
// ---------------------------------------------------------------------------
__device__ __forceinline__ v16h load_a_frag(const _Float16* A, int lda, int m0,
                                            int k0, int lane) {
  const _Float16* row = A + (size_t)(m0 + (lane & 15)) * lda + k0;
  int kh = (lane < 16) ? 0 : 16;
  v16h a;
#pragma unroll
  for (int e = 0; e < 16; ++e) {
    int p = e >> 1, w = e & 1;
    int k = kh + 2 * p + w;
    a[e] = row[k];
  }
  return a;
}

// B stored transposed: BT[N x K]; per-lane fragment = 16 contiguous halves
__device__ __forceinline__ v16h load_b_frag_t(const _Float16* BT, int ldk,
                                              int k0, int n0, int lane) {
  const _Float16* p =
      BT + (size_t)(n0 + (lane & 15)) * ldk + k0 + ((lane < 16) ? 0 : 16);
  v16h b;
#pragma unroll
  for (int e = 0; e < 16; ++e) b[e] = p[e];
  return b;
}

__device__ __forceinline__ v8f wmma_f16(v16h a, v16h b, v8f c) {
  return __builtin_amdgcn_wmma_f32_16x16x32_f16(false, a, false, b, (short)0, c,
                                                false, false);
}

// ---------------------------------------------------------------------------
// f32 -> f16 conversion (plain, and transposing [K x N] -> [N x K])
// ---------------------------------------------------------------------------
__global__ void cvt_f16_kernel(const float* __restrict__ src,
                               _Float16* __restrict__ dst, int n) {
  int i = blockIdx.x * blockDim.x + threadIdx.x;
  if (i < n) dst[i] = (_Float16)src[i];
}

__global__ void cvt_f16_transpose_kernel(const float* __restrict__ src,
                                         _Float16* __restrict__ dst, int K,
                                         int N) {
  int i = blockIdx.x * blockDim.x + threadIdx.x;
  if (i < K * N) {
    int k = i / N, n = i % N;               // coalesced read of src
    dst[(size_t)n * K + k] = (_Float16)src[i];
  }
}

// ---------------------------------------------------------------------------
// WMMA GEMM with transposed B: C[M,N] = relu?(A[M,K] @ BT[N,K]^T + bias)
// grid = (M/16, N/256), block = 256 (8 waves); each wave computes two adjacent
// 16x16 tiles, reusing its A fragment across both.
// ---------------------------------------------------------------------------
__global__ __launch_bounds__(256) void gemm_nt_kernel(
    const _Float16* __restrict__ A, const _Float16* __restrict__ BT,
    const float* __restrict__ bias, float* __restrict__ Cf,
    _Float16* __restrict__ Ch, int M, int N, int K, int relu) {
  int lane = threadIdx.x & 31;
  int wave = threadIdx.x >> 5;
  int m0 = blockIdx.x * 16;
  int n0 = (blockIdx.y * 8 + wave) * 32;

  v8f acc0 = {}, acc1 = {};
  for (int k0 = 0; k0 < K; k0 += 32) {
    if (k0 + 32 < K)
      __builtin_prefetch(A + (size_t)(m0 + (lane & 15)) * K + k0 + 32, 0, 1);
    v16h a = load_a_frag(A, K, m0, k0, lane);
    v16h b0 = load_b_frag_t(BT, K, k0, n0, lane);
    v16h b1 = load_b_frag_t(BT, K, k0, n0 + 16, lane);
    acc0 = wmma_f16(a, b0, acc0);
    acc1 = wmma_f16(a, b1, acc1);
  }

  int mb = m0 + ((lane < 16) ? 0 : 8);
#pragma unroll
  for (int tile = 0; tile < 2; ++tile) {
    int n = n0 + tile * 16 + (lane & 15);
    float bv = bias ? bias[n] : 0.0f;
    const v8f& acc = tile ? acc1 : acc0;
#pragma unroll
    for (int v = 0; v < 8; ++v) {
      float x = acc[v] + bv;
      if (relu) x = fmaxf(x, 0.0f);
      size_t idx = (size_t)(mb + v) * N + n;
      if (Cf) Cf[idx] = x;
      if (Ch) Ch[idx] = (_Float16)x;
    }
  }
}

// ---------------------------------------------------------------------------
// GRU scan: single workgroup, 12 waves. h kept in LDS (f16, double buffered);
// hp = h @ Wh computed in 64-column chunks (3 gates each) via WMMA. Recurrent
// weights pre-transposed: WhT [3H x H].
// ---------------------------------------------------------------------------
__global__ __launch_bounds__(384) void gru_kernel(
    const float* __restrict__ xi,       // [B*T, 3H] input projections (+bias)
    const _Float16* __restrict__ WhT,   // [3H, H] f16 (transposed)
    const float* __restrict__ bhn,      // [H]
    float* __restrict__ c_out,          // [B*T, H] f32 (d_out section)
    _Float16* __restrict__ cP) {        // [B*(T-K), H] f16 compacted rows
  __shared__ _Float16 hA[2][CPC_B][CPC_HCTX];  // 32 KB
  __shared__ float hp[CPC_B][192];             // 12 KB (3 gates x 64 cols)
  int tid = threadIdx.x;
  int lane = tid & 31, wave = tid >> 5;
  int g = wave >> 2, cidx = wave & 3;

  for (int i = tid; i < CPC_B * CPC_HCTX; i += 384)
    (&hA[0][0][0])[i] = (_Float16)0.0f;
  __syncthreads();

  for (int t = 0; t < CPC_T; ++t) {
    int cur = t & 1;
    for (int jc = 0; jc < 8; ++jc) {
      // ---- hp tile: M=16, N=16, K=512 (wave -> gate g, sub-column cidx) ----
      int jloc0 = cidx * 16;
      int n0 = g * CPC_HCTX + jc * 64 + jloc0;
      v8f acc = {};
      const _Float16* hcur = &hA[cur][0][0];
      for (int k0 = 0; k0 < CPC_HCTX; k0 += 32) {
        v16h a = load_a_frag(hcur, CPC_HCTX, 0, k0, lane);
        v16h b = load_b_frag_t(WhT, CPC_HCTX, k0, n0, lane);
        acc = wmma_f16(a, b, acc);
      }
      int col = g * 64 + jloc0 + (lane & 15);
      int bb = (lane < 16) ? 0 : 8;
#pragma unroll
      for (int v = 0; v < 8; ++v) hp[bb + v][col] = acc[v];
      __syncthreads();

      // ---- fused gate nonlinearities for this 16x64 chunk ----
      for (int e = tid; e < CPC_B * 64; e += 384) {
        int b = e >> 6, jl = e & 63;
        int j = jc * 64 + jl;
        const float* xir = xi + (size_t)(b * CPC_T + t) * (3 * CPC_HCTX);
        float xr = xir[j], xu = xir[CPC_HCTX + j], xn = xir[2 * CPC_HCTX + j];
        float hr = hp[b][jl], hu = hp[b][64 + jl], hn = hp[b][128 + jl];
        float r = 1.0f / (1.0f + __expf(-(xr + hr)));
        float u = 1.0f / (1.0f + __expf(-(xu + hu)));
        float nn = tanhf(xn + r * (hn + bhn[j]));
        float hold = (float)hA[cur][b][j];
        float hnew = (1.0f - u) * nn + u * hold;
        hA[1 - cur][b][j] = (_Float16)hnew;
        c_out[(size_t)(b * CPC_T + t) * CPC_HCTX + j] = hnew;
        if (t < CPC_T - CPC_K)
          cP[(size_t)(b * (CPC_T - CPC_K) + t) * CPC_HCTX + j] = (_Float16)hnew;
      }
      __syncthreads();
    }
  }
}

// ---------------------------------------------------------------------------
// Row L2-normalize f32 -> f16 (cols == blockDim.x == 256)
// ---------------------------------------------------------------------------
__global__ __launch_bounds__(256) void normalize_rows_kernel(
    const float* __restrict__ X, _Float16* __restrict__ Y, int cols) {
  __shared__ float red[256];
  size_t base = (size_t)blockIdx.x * cols;
  float v = X[base + threadIdx.x];
  red[threadIdx.x] = v * v;
  __syncthreads();
  for (int s = 128; s > 0; s >>= 1) {
    if ((int)threadIdx.x < s) red[threadIdx.x] += red[threadIdx.x + s];
    __syncthreads();
  }
  float inv = rsqrtf(red[0] + 1e-30f);
  Y[base + threadIdx.x] = (_Float16)(v * inv);
}

__global__ void zero_loss_kernel(float* p) { *p = 0.0f; }

// ---------------------------------------------------------------------------
// Fused InfoNCE: grid = (125, 12), 4 waves/block; each wave owns one 16-row
// tile of p and streams all 500 column tiles of zn_k with online logsumexp.
// ---------------------------------------------------------------------------
__global__ __launch_bounds__(128) void infonce_kernel(
    const _Float16* __restrict__ P,    // [8000,256] normalized
    const _Float16* __restrict__ ZN,   // [8192,256] normalized (row b*T+t)
    float* __restrict__ loss) {
  const int TCOL = CPC_T - CPC_K;      // 500 tiles of 16
  int lane = threadIdx.x & 31;
  int wave = threadIdx.x >> 5;
  int rowTile = blockIdx.x * 4 + wave;
  int k = blockIdx.y + 1;
  if (rowTile >= TCOL) return;
  int m0 = rowTile * 16;

  v16h afr[8];
#pragma unroll
  for (int kk = 0; kk < 8; ++kk)
    afr[kk] = load_a_frag(P, CPC_DLAT, m0, kk * 32, lane);

  float runm[8], runs[8], diag[8];
#pragma unroll
  for (int v = 0; v < 8; ++v) { runm[v] = -1e30f; runs[v] = 0.0f; diag[v] = 0.0f; }

  int ncol = lane & 15;
  int mbase = m0 + ((lane < 16) ? 0 : 8);
  int koff = (lane < 16) ? 0 : 16;

  for (int jt = 0; jt < TCOL; ++jt) {
    int j = jt * 16 + ncol;                          // global column (row of zt)
    int zrow = (j / TCOL) * CPC_T + k + (j % TCOL);  // shifted z row
    const _Float16* zr = ZN + (size_t)zrow * CPC_DLAT;
    v8f acc = {};
#pragma unroll
    for (int kk = 0; kk < 8; ++kk) {
      v16h b;
#pragma unroll
      for (int e = 0; e < 16; ++e) b[e] = zr[kk * 32 + koff + e];
      acc = wmma_f16(afr[kk], b, acc);
    }
#pragma unroll
    for (int v = 0; v < 8; ++v) {
      float s = acc[v] * 10.0f;                      // 1 / TEMP
      if (j == mbase + v) diag[v] = s;               // diagonal capture
      float nm = fmaxf(runm[v], s);
      runs[v] = runs[v] * __expf(runm[v] - nm) + __expf(s - nm);
      runm[v] = nm;
    }
  }

  // combine the 16 lanes that share each row (butterfly over masks 1,2,4,8)
#pragma unroll
  for (int off = 1; off < 16; off <<= 1) {
#pragma unroll
    for (int v = 0; v < 8; ++v) {
      float om = __shfl_xor(runm[v], off, 32);
      float os = __shfl_xor(runs[v], off, 32);
      float nm = fmaxf(runm[v], om);
      runs[v] = runs[v] * __expf(runm[v] - nm) + os * __expf(om - nm);
      runm[v] = nm;
      diag[v] += __shfl_xor(diag[v], off, 32);
    }
  }
  if ((lane & 15) == 0) {
    float part = 0.0f;
#pragma unroll
    for (int v = 0; v < 8; ++v) part += (runm[v] + __logf(runs[v])) - diag[v];
    atomicAdd(loss, part * (1.0f / ((float)CPC_N * (float)CPC_K)));
  }
}

// ---------------------------------------------------------------------------
extern "C" void kernel_launch(void* const* d_in, const int* in_sizes, int n_in,
                              void* d_out, int out_size, void* d_ws,
                              size_t ws_size, hipStream_t stream) {
  const float* x       = (const float*)d_in[0];
  const float* enc_w1  = (const float*)d_in[1];
  const float* enc_b1  = (const float*)d_in[2];
  const float* enc_w2  = (const float*)d_in[3];
  const float* enc_b2  = (const float*)d_in[4];
  const float* enc_w3  = (const float*)d_in[5];
  const float* enc_b3  = (const float*)d_in[6];
  const float* gru_wi  = (const float*)d_in[7];
  const float* gru_bi  = (const float*)d_in[8];
  const float* gru_wh  = (const float*)d_in[9];
  const float* gru_bhn = (const float*)d_in[10];
  const float* pred_w1 = (const float*)d_in[11];
  const float* pred_b1 = (const float*)d_in[12];
  const float* pred_w2 = (const float*)d_in[13];
  const float* pred_b2 = (const float*)d_in[14];

  float* z_out = (float*)d_out;                 // [B*T, 256]
  float* c_out = z_out + Z_ELEMS;               // [B*T, 512]
  float* loss  = z_out + Z_ELEMS + C_ELEMS;     // scalar

  // workspace carving (256B aligned)
  char* w = (char*)d_ws;
  auto carve = [&](size_t bytes) {
    void* p = (void*)w;
    w += (bytes + 255) & ~(size_t)255;
    return p;
  };
  const int ROWS = CPC_B * CPC_T;  // 8192
  _Float16* Xh   = (_Float16*)carve((size_t)ROWS * CPC_DIN * 2);
  _Float16* W1t  = (_Float16*)carve((size_t)CPC_DIN * CPC_HENC * 2);
  _Float16* W2t  = (_Float16*)carve((size_t)CPC_HENC * CPC_HENC * 2);
  _Float16* W3t  = (_Float16*)carve((size_t)CPC_HENC * CPC_DLAT * 2);
  _Float16* WGIt = (_Float16*)carve((size_t)CPC_DLAT * 3 * CPC_HCTX * 2);
  _Float16* WGHt = (_Float16*)carve((size_t)CPC_HCTX * 3 * CPC_HCTX * 2);
  _Float16* WP1t = (_Float16*)carve((size_t)CPC_HCTX * CPC_HPRED * 2);
  _Float16* WP2t = (_Float16*)carve((size_t)CPC_HPRED * CPC_DLAT * 2);
  _Float16* H1h  = (_Float16*)carve((size_t)ROWS * CPC_HENC * 2);
  _Float16* H2h  = (_Float16*)carve((size_t)ROWS * CPC_HENC * 2);
  _Float16* Zh   = (_Float16*)carve((size_t)ROWS * CPC_DLAT * 2);
  float*    XI   = (float*)carve((size_t)ROWS * 3 * CPC_HCTX * 4);
  _Float16* CPh  = (_Float16*)carve((size_t)CPC_N * CPC_HCTX * 2);
  _Float16* PH1  = (_Float16*)carve((size_t)CPC_N * CPC_HPRED * 2);
  float*    Pf   = (float*)carve((size_t)CPC_N * CPC_DLAT * 4);
  _Float16* Ph   = (_Float16*)carve((size_t)CPC_N * CPC_DLAT * 2);
  _Float16* ZNh  = (_Float16*)carve((size_t)ROWS * CPC_DLAT * 2);

  auto cvt = [&](const float* s, _Float16* d, int n) {
    cvt_f16_kernel<<<(n + 255) / 256, 256, 0, stream>>>(s, d, n);
  };
  auto cvtT = [&](const float* s, _Float16* d, int K, int N) {
    int n = K * N;
    cvt_f16_transpose_kernel<<<(n + 255) / 256, 256, 0, stream>>>(s, d, K, N);
  };
  cvt(x, Xh, ROWS * CPC_DIN);
  cvtT(enc_w1, W1t, CPC_DIN, CPC_HENC);
  cvtT(enc_w2, W2t, CPC_HENC, CPC_HENC);
  cvtT(enc_w3, W3t, CPC_HENC, CPC_DLAT);
  cvtT(gru_wi, WGIt, CPC_DLAT, 3 * CPC_HCTX);
  cvtT(gru_wh, WGHt, CPC_HCTX, 3 * CPC_HCTX);
  cvtT(pred_w1, WP1t, CPC_HCTX, CPC_HPRED);
  cvtT(pred_w2, WP2t, CPC_HPRED, CPC_DLAT);

  auto gemm = [&](const _Float16* A, const _Float16* BT, const float* bias,
                  float* Cf, _Float16* Ch, int M, int N, int K, int relu) {
    gemm_nt_kernel<<<dim3(M / 16, N / 256), 256, 0, stream>>>(A, BT, bias, Cf,
                                                              Ch, M, N, K, relu);
  };
  // encoder
  gemm(Xh, W1t, enc_b1, nullptr, H1h, ROWS, CPC_HENC, CPC_DIN, 1);
  gemm(H1h, W2t, enc_b2, nullptr, H2h, ROWS, CPC_HENC, CPC_HENC, 1);
  gemm(H2h, W3t, enc_b3, z_out, Zh, ROWS, CPC_DLAT, CPC_HENC, 0);
  // GRU input projections + scan
  gemm(Zh, WGIt, gru_bi, XI, nullptr, ROWS, 3 * CPC_HCTX, CPC_DLAT, 0);
  gru_kernel<<<1, 384, 0, stream>>>(XI, WGHt, gru_bhn, c_out, CPh);
  // prediction net
  gemm(CPh, WP1t, pred_b1, nullptr, PH1, CPC_N, CPC_HPRED, CPC_HCTX, 1);
  gemm(PH1, WP2t, pred_b2, Pf, nullptr, CPC_N, CPC_DLAT, CPC_HPRED, 0);
  // normalize p and z rows to f16
  normalize_rows_kernel<<<CPC_N, 256, 0, stream>>>(Pf, Ph, CPC_DLAT);
  normalize_rows_kernel<<<ROWS, 256, 0, stream>>>(z_out, ZNh, CPC_DLAT);
  // fused InfoNCE loss
  zero_loss_kernel<<<1, 1, 0, stream>>>(loss);
  infonce_kernel<<<dim3((CPC_T - CPC_K) / 4, CPC_K), 128, 0, stream>>>(Ph, ZNh,
                                                                      loss);
}
